// model_16209206575474
// MI455X (gfx1250) — matrix operand order repaired and checked by reading, not compiled, not run on hardware
//
#include <hip/hip_runtime.h>
#include <hip/hip_bf16.h>
#include <math.h>

// ---------- types for WMMA ----------
typedef __attribute__((ext_vector_type(16))) __bf16 v16bf;
typedef __attribute__((ext_vector_type(8)))  float  v8f;

union V16 { v16bf v; unsigned short s[16]; uint4 u[2]; };

__device__ __forceinline__ unsigned short f2bf(float f) {
  unsigned u = __float_as_uint(f);
  u += 0x7FFFu + ((u >> 16) & 1u);       // round-to-nearest-even
  return (unsigned short)(u >> 16);
}

#define Bsz 32
#define Dsz 256
#define Tsz 4096
#define Ksz 1024
#define Nsz (Bsz * Tsz)          // 131072
#define PAD 264                  // LDS row stride (bf16 units) to break bank conflicts

// ---------------- kernel 0: zero histogram + loss accumulator ----------------
__global__ void init_ws_kernel(int* __restrict__ hist, float* __restrict__ lossSum) {
  int i = threadIdx.x;
  if (i < Ksz) hist[i] = 0;
  if (i == 0) *lossSum = 0.0f;
}

// ---------------- kernel 1: codebook fp32->bf16 + ||e_k||^2 ----------------
__global__ void prep_kernel(const float* __restrict__ cb,
                            unsigned short* __restrict__ cbbf,
                            float* __restrict__ enorm) {
  __shared__ float red[256];
  const int k = blockIdx.x;
  float v = cb[(size_t)k * Dsz + threadIdx.x];
  cbbf[(size_t)k * Dsz + threadIdx.x] = f2bf(v);
  red[threadIdx.x] = v * v;
  __syncthreads();
  for (int off = 128; off > 0; off >>= 1) {
    if (threadIdx.x < (unsigned)off) red[threadIdx.x] += red[threadIdx.x + off];
    __syncthreads();
  }
  if (threadIdx.x == 0) enorm[k] = red[0];
}

// ---- async staging: 128 codes x 256 dims bf16 (64 KB) -> LDS, padded rows ----
// Each thread issues 16x global_load_async_to_lds_b128 (ASYNCcnt-tracked).
__device__ __forceinline__ void async_stage_chunk(const unsigned short* __restrict__ src,
                                                  unsigned ldsBase, int tid) {
#pragma unroll
  for (int i = 0; i < 16; ++i) {
    int idx  = i * 256 + tid;        // 4096 16-byte chunks total
    int code = idx >> 5;             // 32 chunks per code row (512 B)
    int c16  = idx & 31;
    unsigned voff  = (unsigned)(code * 512 + c16 * 16);          // global byte offset
    unsigned laddr = ldsBase + (unsigned)(code * (PAD * 2) + c16 * 16); // LDS byte addr
    asm volatile("global_load_async_to_lds_b128 %0, %1, %2"
                 :: "v"(laddr), "v"(voff), "s"(src)
                 : "memory");
  }
}

// ---------------- kernel 2: WMMA distance + argmin ----------------
// grid (T/256, B), block 256 (8 waves). Wave w owns 32 points (two 16-pt A tiles).
__global__ __launch_bounds__(256) void vq_argmin_kernel(
    const float* __restrict__ x,              // (32,256,4096)  [b,d,t]
    const unsigned short* __restrict__ cbbf,  // (1024,256) bf16
    const float* __restrict__ enorm,          // (1024)
    int* __restrict__ idxOut,                 // (131072)
    int* __restrict__ hist)                   // (1024)
{
  __shared__ unsigned short codelds[2][128 * PAD];  // double-buffered code chunks
  __shared__ float enlds[Ksz];

  const int tid  = threadIdx.x;
  const int lane = tid & 31;
  const int wave = tid >> 5;
  const int m    = lane & 15;     // point slot for A-load; code column for C
  const int sel  = lane >> 4;     // half-wave selector
  const int b    = blockIdx.y;
  const int t0   = blockIdx.x * 256 + wave * 32;

  const unsigned ldsBase0 = (unsigned)(size_t)&codelds[0][0];
  const unsigned ldsBase1 = (unsigned)(size_t)&codelds[1][0];

  for (int i = tid; i < Ksz; i += 256) enlds[i] = enorm[i];

  // ---- load two A tiles (2 x 16 points x 256 dims) into registers, WMMA A-striping ----
  // lanes 0-15 hold K [k0..k0+15]; lanes 16-31 hold [k0+16..k0+31]
  const float* xp0 = x + ((size_t)b * Dsz) * Tsz + (t0 + m);
  const float* xp1 = xp0 + 16;
  V16 A0[8], A1[8];
#pragma unroll
  for (int ks = 0; ks < 8; ++ks) {
    const int k0 = ks * 32;
#pragma unroll
    for (int j = 0; j < 8; ++j) {
      A0[ks].s[j]     = f2bf(xp0[(size_t)(k0 + sel * 16 + j) * Tsz]);
      A0[ks].s[8 + j] = f2bf(xp0[(size_t)(k0 + sel * 16 + 8 + j) * Tsz]);
      A1[ks].s[j]     = f2bf(xp1[(size_t)(k0 + sel * 16 + j) * Tsz]);
      A1[ks].s[8 + j] = f2bf(xp1[(size_t)(k0 + sel * 16 + 8 + j) * Tsz]);
    }
  }

  float bestd[2][8];
  int   besti[2][8];
#pragma unroll
  for (int r = 0; r < 8; ++r) {
    bestd[0][r] = 3.4e38f; besti[0][r] = 0;
    bestd[1][r] = 3.4e38f; besti[1][r] = 0;
  }

  // prologue: stage chunk 0 into buffer 0
  async_stage_chunk(cbbf, ldsBase0, tid);

  for (int chunk = 0; chunk < 8; ++chunk) {
    // my chunk-`chunk` async writes done; barrier => everyone's done + all
    // waves finished reading buffer (chunk+1)&1 in iteration chunk-1.
    asm volatile("s_wait_asynccnt 0x0" ::: "memory");
    __syncthreads();
    if (chunk < 7) {
      unsigned dstBase = ((chunk + 1) & 1) ? ldsBase1 : ldsBase0;
      async_stage_chunk(cbbf + (size_t)(chunk + 1) * 128 * Dsz, dstBase, tid);
    }
    const unsigned short* buf = &codelds[chunk & 1][0];

    for (int ct = 0; ct < 8; ++ct) {
      const int code = chunk * 128 + ct * 16 + m;   // C: lane&15 = code column
      const float en = enlds[code];                 // issue early, hide under WMMAs

      v8f acc0 = (v8f)0.0f;
      v8f acc1 = (v8f)0.0f;
      // B striping: lane holds dims [k0 + sel*16 .. +15] of code column (lane&15)
      const unsigned short* bp = buf + (ct * 16 + m) * PAD + sel * 16;

      // register double-buffer for B fragments: load ks+1 before consuming ks
      V16 Bcur, Bnxt;
      Bcur.u[0] = *(const uint4*)(bp);
      Bcur.u[1] = *(const uint4*)(bp + 8);
#pragma unroll
      for (int ks = 0; ks < 8; ++ks) {
        if (ks < 7) {
          Bnxt.u[0] = *(const uint4*)(bp + (ks + 1) * 32);
          Bnxt.u[1] = *(const uint4*)(bp + (ks + 1) * 32 + 8);
        }
        acc0 = __builtin_amdgcn_wmma_f32_16x16x32_bf16(
            false, A0[ks].v, false, Bcur.v, (short)0, acc0, false, false);
        acc1 = __builtin_amdgcn_wmma_f32_16x16x32_bf16(
            false, A1[ks].v, false, Bcur.v, (short)0, acc1, false, false);
        if (ks < 7) { Bcur = Bnxt; }
      }

#pragma unroll
      for (int r = 0; r < 8; ++r) {
        // dist = ||e||^2 - 2 x.e   (||x||^2 constant per point -> dropped)
        float d0 = fmaf(-2.0f, acc0[r], en);
        float d1 = fmaf(-2.0f, acc1[r], en);
        if (d0 < bestd[0][r]) { bestd[0][r] = d0; besti[0][r] = code; }
        if (d1 < bestd[1][r]) { bestd[1][r] = d1; besti[1][r] = code; }
      }
    }
  }

  // reduce argmin across the 16 lanes of each half (C: VGPR r -> point r + 8*sel)
#pragma unroll
  for (int g = 0; g < 2; ++g) {
#pragma unroll
    for (int r = 0; r < 8; ++r) {
      float bd = bestd[g][r];
      int   bi = besti[g][r];
#pragma unroll
      for (int off = 8; off >= 1; off >>= 1) {
        float od = __shfl_xor(bd, off, 32);
        int   oi = __shfl_xor(bi, off, 32);
        if (od < bd || (od == bd && oi < bi)) { bd = od; bi = oi; }
      }
      if (m == 0) {
        int n = b * Tsz + t0 + g * 16 + r + 8 * sel;
        idxOut[n] = bi;
        atomicAdd(&hist[bi], 1);
      }
    }
  }
}

// ---------------- kernel 3: gather quantized + commitment loss ----------------
__global__ __launch_bounds__(256) void gather_kernel(
    const float* __restrict__ x, const float* __restrict__ cb,
    const int* __restrict__ idx, float* __restrict__ outq,
    float* __restrict__ lossSum)
{
  __shared__ float red[256];
  const int b = blockIdx.y;
  const int t = blockIdx.x * 256 + threadIdx.x;
  const int id = idx[b * Tsz + t];
  const float4* crow = (const float4*)(cb + (size_t)id * Dsz);
  const float* xb = x    + (size_t)b * Dsz * Tsz + t;
  float*       ob = outq + (size_t)b * Dsz * Tsz + t;

  float acc = 0.0f;
  for (int d4 = 0; d4 < 64; ++d4) {
    float4 q = crow[d4];
    float x0 = xb[(size_t)(d4 * 4 + 0) * Tsz];
    float x1 = xb[(size_t)(d4 * 4 + 1) * Tsz];
    float x2 = xb[(size_t)(d4 * 4 + 2) * Tsz];
    float x3 = xb[(size_t)(d4 * 4 + 3) * Tsz];
    ob[(size_t)(d4 * 4 + 0) * Tsz] = q.x;
    ob[(size_t)(d4 * 4 + 1) * Tsz] = q.y;
    ob[(size_t)(d4 * 4 + 2) * Tsz] = q.z;
    ob[(size_t)(d4 * 4 + 3) * Tsz] = q.w;
    acc = fmaf(q.x - x0, q.x - x0, acc);
    acc = fmaf(q.y - x1, q.y - x1, acc);
    acc = fmaf(q.z - x2, q.z - x2, acc);
    acc = fmaf(q.w - x3, q.w - x3, acc);
  }
  red[threadIdx.x] = acc;
  __syncthreads();
  for (int off = 128; off > 0; off >>= 1) {
    if (threadIdx.x < (unsigned)off) red[threadIdx.x] += red[threadIdx.x + off];
    __syncthreads();
  }
  if (threadIdx.x == 0) atomicAdd(lossSum, red[0]);
}

// ---------------- kernel 4: finalize loss + perplexity ----------------
__global__ void finalize_kernel(const int* __restrict__ hist,
                                const float* __restrict__ lossSum,
                                float* __restrict__ out_loss,
                                float* __restrict__ out_perp)
{
  __shared__ float red[1024];
  const int k = threadIdx.x;   // 1024 threads
  float p = (float)hist[k] * (1.0f / (float)Nsz);
  red[k] = p * logf(p + 1e-10f);
  __syncthreads();
  for (int off = 512; off > 0; off >>= 1) {
    if (k < off) red[k] += red[k + off];
    __syncthreads();
  }
  if (k == 0) {
    *out_perp = expf(-red[0]);
    *out_loss = 0.25f * (*lossSum) / ((float)Nsz * (float)Dsz);
  }
}

// ---------------- launch ----------------
extern "C" void kernel_launch(void* const* d_in, const int* in_sizes, int n_in,
                              void* d_out, int out_size, void* d_ws, size_t ws_size,
                              hipStream_t stream) {
  const float* x  = (const float*)d_in[0];   // (32,256,4096) fp32
  const float* cb = (const float*)d_in[1];   // (1024,256)    fp32

  float* out = (float*)d_out;
  const size_t Q = (size_t)Bsz * Dsz * Tsz;  // 33554432
  float* out_loss = out;
  float* out_q    = out + 1;
  float* out_perp = out + 1 + Q;
  int*   out_idx  = (int*)(out + 2 + Q);     // idx as int32 bits

  unsigned short* cbbf = (unsigned short*)d_ws;             // 512 KB bf16 codebook
  float* enorm   = (float*)((char*)d_ws + 524288);          // 4 KB
  int*   hist    = (int*)((char*)d_ws + 528384);            // 4 KB
  float* lossSum = (float*)((char*)d_ws + 532480);          // 4 B

  init_ws_kernel<<<1, 1024, 0, stream>>>(hist, lossSum);
  prep_kernel<<<Ksz, 256, 0, stream>>>(cb, cbbf, enorm);
  vq_argmin_kernel<<<dim3(Tsz / 256, Bsz), 256, 0, stream>>>(x, cbbf, enorm, out_idx, hist);
  gather_kernel<<<dim3(Tsz / 256, Bsz), 256, 0, stream>>>(x, cb, out_idx, out_q, lossSum);
  finalize_kernel<<<1, 1024, 0, stream>>>(hist, lossSum, out_loss, out_perp);
}
